// QuantizedAttention_81458349736284
// MI455X (gfx1250) — matrix operand order repaired and checked by reading, not compiled, not run on hardware
//
#include <hip/hip_runtime.h>

// ---------------------------------------------------------------------------
// Quantized causal attention block for MI455X (gfx1250, wave32, WMMA).
// Compute-bound (~103 GFLOP vs ~5us of HBM traffic at 23.3 TB/s), so every
// GEMM-shaped op goes through v_wmma_f32_16x16x32_bf16, and the bf16 K/V tile
// staging uses gfx1250 async global->LDS copies (ASYNCcnt).
// ---------------------------------------------------------------------------

typedef __attribute__((ext_vector_type(16))) __bf16 vbf16;
typedef __attribute__((ext_vector_type(8)))  float  v8f;

#define WMMA_BF16(a, b, c) \
  __builtin_amdgcn_wmma_f32_16x16x32_bf16(false, (a), false, (b), (short)0, (c), false, false)

static constexpr int Bz = 2, S = 2048, D = 1024, H = 16, HD = 64, E3 = 3072;
static constexpr int MROWS = Bz * S;            // 4096
static constexpr long NELEM = (long)Bz * S * D; // 4,194,304

// ----- WMMA fragment loaders (this file's contiguous k order k = 16*(lane/16) + s, same in A and B) ---
// A 16x32 bf16: lane L (L<16) row M=L holds K 0..15; lane L+16 holds
// K 16..31; 2 halves per VGPR.
__device__ __forceinline__ vbf16 load_a_frag(const __bf16* base, int ld) {
  int lane = threadIdx.x & 31;
  int row  = lane & 15;
  int kh   = (lane >> 4) << 4;       // 0 or 16
  const __bf16* p = base + row * ld;
  vbf16 a;
#pragma unroll
  for (int v = 0; v < 4; ++v) {
    a[2 * v]     = p[kh + 2 * v];
    a[2 * v + 1] = p[kh + 2 * v + 1];
    a[2 * v + 8] = p[8 + kh + 2 * v];
    a[2 * v + 9] = p[8 + kh + 2 * v + 1];
  }
  return a;
}

// B 32x16 bf16: column N = lane&15; lanes 0-15 hold K=0..15, lanes 16-31 hold
// K=16..31. Tile stored as [n][k], so the 16 K-values per lane are contiguous.
__device__ __forceinline__ vbf16 load_b_frag(const __bf16* base, int ld) {
  int lane = threadIdx.x & 31;
  const __bf16* p = base + (lane & 15) * ld + ((lane >> 4) << 4);
  vbf16 b;
#pragma unroll
  for (int i = 0; i < 16; ++i) b[i] = p[i];
  return b;
}

// Same B fragment but from a [k][n] row-major tile (strided gather).
__device__ __forceinline__ vbf16 load_b_fragT(const __bf16* base, int ld) {
  int lane = threadIdx.x & 31;
  int n  = lane & 15;
  int kh = (lane >> 4) << 4;
  vbf16 b;
#pragma unroll
  for (int i = 0; i < 16; ++i) b[i] = base[(kh + i) * ld + n];
  return b;
}

// C/D 16x16 f32: N = lane&15; VGPR r -> M = r + 8*(lane>=16).

// ---------------------------------------------------------------------------
// Tiled GEMM: C[M,N] = A[M,K] * W[N,K]^T + bias[N], fp32 in/out, bf16 WMMA.
// 128x64 workgroup tile, 4 waves; each wave does 32x64 = 2x4 WMMA tiles per
// K-step of 32 (8 WMMAs per barrier interval, 2 A-frags reused over 4 B-frags).
// ---------------------------------------------------------------------------
__global__ __launch_bounds__(128) void gemm_bf16_wmma(
    const float* __restrict__ A, const float* __restrict__ W,
    const float* __restrict__ bias, float* __restrict__ C,
    int M, int N, int K) {
  constexpr int TM = 128, TN = 64, KS = 32, LD = KS + 8;
  __shared__ __bf16 As[TM][LD];
  __shared__ __bf16 Ws[TN][LD];

  int tm = blockIdx.y * TM;
  int tn = blockIdx.x * TN;
  int wave = threadIdx.x >> 5;
  int lane = threadIdx.x & 31;

  v8f acc[2][4] = {};
  for (int k0 = 0; k0 < K; k0 += KS) {
    if (k0 + KS < K) {
      __builtin_prefetch(&A[(long)(tm + threadIdx.x) * K + k0 + KS], 0, 0);
      __builtin_prefetch(&W[(long)(tn + (threadIdx.x >> 1)) * K + k0 + KS], 0, 0);
    }
    for (int i = threadIdx.x; i < TM * KS; i += 128) {
      int r = i >> 5, c = i & 31;
      As[r][c] = (__bf16)A[(long)(tm + r) * K + k0 + c];
      if (r < TN) Ws[r][c] = (__bf16)W[(long)(tn + r) * K + k0 + c];
    }
    __syncthreads();
    vbf16 a0 = load_a_frag(&As[wave * 32][0], LD);
    vbf16 a1 = load_a_frag(&As[wave * 32 + 16][0], LD);
#pragma unroll
    for (int nt = 0; nt < 4; ++nt) {
      vbf16 b = load_b_frag(&Ws[nt * 16][0], LD);
      acc[0][nt] = WMMA_BF16(a0, b, acc[0][nt]);
      acc[1][nt] = WMMA_BF16(a1, b, acc[1][nt]);
    }
    __syncthreads();
  }
  int n  = lane & 15;
  int mo = (lane >> 4) << 3;
#pragma unroll
  for (int mt = 0; mt < 2; ++mt)
#pragma unroll
    for (int nt = 0; nt < 4; ++nt) {
      int col = tn + nt * 16 + n;
      float bv = bias[col];
#pragma unroll
      for (int r = 0; r < 8; ++r) {
        int row = tm + wave * 32 + mt * 16 + mo + r;
        C[(long)row * N + col] = acc[mt][nt][r] + bv;
      }
    }
}

// ---------------------------------------------------------------------------
__global__ void init_amax_kernel(unsigned* a) {
  if (threadIdx.x < 3) a[threadIdx.x] = 0u;
}

// max |x| over a strided 2-D region, grid-stride + wave reduce + atomicMax on
// float bits (valid since values are >= 0).
__global__ void amax2d_kernel(const float* __restrict__ base, int rowstride,
                              int cols, int total, unsigned* slot) {
  float m = 0.f;
  for (int i = blockIdx.x * blockDim.x + threadIdx.x; i < total;
       i += gridDim.x * blockDim.x) {
    int r = i / cols, c = i - r * cols;
    m = fmaxf(m, fabsf(base[(long)r * rowstride + c]));
  }
#pragma unroll
  for (int off = 16; off; off >>= 1) m = fmaxf(m, __shfl_xor(m, off, 32));
  __shared__ float sm[8];
  int wave = threadIdx.x >> 5, lane = threadIdx.x & 31;
  if (lane == 0) sm[wave] = m;
  __syncthreads();
  if (threadIdx.x == 0) {
    float mm = sm[0];
    for (int w = 1; w < (int)(blockDim.x >> 5); ++w) mm = fmaxf(mm, sm[w]);
    atomicMax(slot, __float_as_uint(mm));
  }
}

// fake-quantize K/V, emit the `present` outputs ([B,H,S,hd] fp32) and bf16
// copies laid out per-head for the attention kernel.
__global__ void quant_kv_kernel(const float* __restrict__ qkv,
                                const unsigned* __restrict__ amax,
                                float* __restrict__ pk, float* __restrict__ pv,
                                __bf16* __restrict__ kbf, __bf16* __restrict__ vbf) {
  long i = (long)blockIdx.x * blockDim.x + threadIdx.x;
  if (i >= NELEM) return;
  int m = (int)(i >> 10);          // b*S + s
  int j = (int)(i & 1023);         // column within D
  int b = m >> 11, s = m & 2047;
  int h = j >> 6, dd = j & 63;
  long o = (((long)(b * H + h) * S + s) << 6) + dd;
  float sk = __uint_as_float(amax[0]) * (1.0f / 127.0f);
  float sv = __uint_as_float(amax[1]) * (1.0f / 127.0f);
  float kv = qkv[(long)m * E3 + D + j];
  float vv = qkv[(long)m * E3 + 2 * D + j];
  float kq = fminf(fmaxf(rintf(kv / sk), -128.f), 127.f) * sk;
  float vq = fminf(fmaxf(rintf(vv / sv), -128.f), 127.f) * sv;
  pk[o] = kq;  pv[o] = vq;
  kbf[o] = (__bf16)kq;  vbf[o] = (__bf16)vq;
}

// ---------------------------------------------------------------------------
// Flash attention: one block = one (b,h) and 64 Q rows; 4 waves * 16 rows.
// K/V tiles staged in LDS via gfx1250 async global->LDS B128 copies
// (ASYNCcnt-tracked, s_wait_asynccnt before the barrier). Causal bound cuts
// the K loop; scores: 4 WMMA; P@V: 4 WMMA per tile.
// ---------------------------------------------------------------------------
__global__ __launch_bounds__(128) void attn_kernel(
    const float* __restrict__ qkv, const __bf16* __restrict__ kbf,
    const __bf16* __restrict__ vbf, float* __restrict__ attn_out,
    unsigned* amax_slot) {
  constexpr int QT = 64, KT = 32, LDQ = HD + 8;
  __shared__ __bf16 Qs[QT][LDQ];
  __shared__ __bf16 Ks[KT][LDQ];   // [krow][hd]
  __shared__ __bf16 Vs[KT][LDQ];   // [krow][hd]
  __shared__ float  Ss[4][16][KT + 4];
  __shared__ __bf16 Ps[4][16][KT + 8];
  __shared__ float  Alpha[4][16];
  __shared__ float  Lsum[4][16];

  int bh = blockIdx.y;
  int b = bh >> 4, h = bh & 15;
  int q0 = blockIdx.x * QT;
  int wave = threadIdx.x >> 5, lane = threadIdx.x & 31;
  int n = lane & 15, mo = (lane >> 4) << 3;
  long kvbase = ((long)bh * S) << 6;        // head-major [B*H][S][64], elems

  for (int i = threadIdx.x; i < QT * HD; i += 128) {
    int r = i >> 6, c = i & 63;
    Qs[r][c] = (__bf16)qkv[(long)(b * S + q0 + r) * E3 + h * HD + c];
  }
  __syncthreads();

  // LDS byte addresses (generic shared pointers keep the LDS offset in the
  // low 32 bits) for the async-copy destination VGPRs.
  unsigned ksbase = (unsigned)(unsigned long long)&Ks[0][0];
  unsigned vsbase = (unsigned)(unsigned long long)&Vs[0][0];
  const char* kg = (const char*)(kbf + kvbase);
  const char* vg = (const char*)(vbf + kvbase);

  v8f acc_o[4] = {};
  float m_run = -3.0e38f, l_run = 0.0f;

  int niter = (q0 + QT) / KT;               // causal bound
  for (int it = 0; it < niter; ++it) {
    int kt = it * KT;
    // Async global->LDS: 256 granules of 16B per matrix (KT*HD bf16), padded
    // LDS rows handled by per-granule dest addresses.
#pragma unroll
    for (int gi = 0; gi < 2; ++gi) {
      int g = threadIdx.x + gi * 128;       // granule id, 8 bf16 each
      int r = g >> 3, c = (g & 7) << 3;
      unsigned loff = (unsigned)((r * LDQ + c) * 2);
      unsigned long long kga =
          (unsigned long long)(kg + ((long)(kt + r) * HD + c) * 2);
      unsigned long long vga =
          (unsigned long long)(vg + ((long)(kt + r) * HD + c) * 2);
      asm volatile("global_load_async_to_lds_b128 %0, %1, off"
                   :: "v"(ksbase + loff), "v"(kga) : "memory");
      asm volatile("global_load_async_to_lds_b128 %0, %1, off"
                   :: "v"(vsbase + loff), "v"(vga) : "memory");
    }
    asm volatile("s_wait_asynccnt 0x0" ::: "memory");
    __syncthreads();

    // scores = Q * K^T over hd=64 (two K-steps of 32)
    vbf16 aq0 = load_a_frag(&Qs[wave * 16][0], LDQ);
    vbf16 aq1 = load_a_frag(&Qs[wave * 16][32], LDQ);
    v8f sc[2] = {};
#pragma unroll
    for (int nt = 0; nt < 2; ++nt) {
      vbf16 b0 = load_b_frag(&Ks[nt * 16][0], LDQ);
      sc[nt] = WMMA_BF16(aq0, b0, sc[nt]);
      vbf16 b1 = load_b_frag(&Ks[nt * 16][32], LDQ);
      sc[nt] = WMMA_BF16(aq1, b1, sc[nt]);
    }
#pragma unroll
    for (int nt = 0; nt < 2; ++nt)
#pragma unroll
      for (int r = 0; r < 8; ++r) {
        int row = mo + r, col = nt * 16 + n;
        int qrow = q0 + wave * 16 + row, krow = kt + col;
        Ss[wave][row][col] =
            sc[nt][r] * 0.125f + (krow > qrow ? -1.0e9f : 0.0f);
      }
    __syncthreads();

    // online softmax: lanes 0-15 each own one of the wave's 16 rows
    if (lane < 16) {
      int row = lane;
      float tmax = -3.0e38f;
#pragma unroll
      for (int j = 0; j < KT; ++j) tmax = fmaxf(tmax, Ss[wave][row][j]);
      float m_new = fmaxf(m_run, tmax);
      float alpha = __expf(m_run - m_new);
      float psum = 0.f;
#pragma unroll
      for (int j = 0; j < KT; ++j) {
        float p = __expf(Ss[wave][row][j] - m_new);
        psum += p;
        Ps[wave][row][j] = (__bf16)p;
      }
      l_run = l_run * alpha + psum;
      m_run = m_new;
      Alpha[wave][row] = alpha;
    }
    __syncthreads();

    // rescale running O, then O += P * V
    float al[8];
#pragma unroll
    for (int r = 0; r < 8; ++r) al[r] = Alpha[wave][mo + r];
    vbf16 ap = load_a_frag(&Ps[wave][0][0], KT + 8);
#pragma unroll
    for (int nt = 0; nt < 4; ++nt) {
#pragma unroll
      for (int r = 0; r < 8; ++r) acc_o[nt][r] *= al[r];
      vbf16 bv = load_b_fragT(&Vs[0][nt * 16], LDQ);
      acc_o[nt] = WMMA_BF16(ap, bv, acc_o[nt]);
    }
    __syncthreads();
  }

  if (lane < 16) Lsum[wave][lane] = l_run;
  __syncthreads();
  float rl[8];
#pragma unroll
  for (int r = 0; r < 8; ++r) rl[r] = 1.0f / Lsum[wave][mo + r];
  float lm = 0.f;
#pragma unroll
  for (int nt = 0; nt < 4; ++nt)
#pragma unroll
    for (int r = 0; r < 8; ++r) {
      float val = acc_o[nt][r] * rl[r];
      int qrow = q0 + wave * 16 + mo + r;
      attn_out[(long)(b * S + qrow) * D + h * HD + nt * 16 + n] = val;
      lm = fmaxf(lm, fabsf(val));
    }
  atomicMax(amax_slot, __float_as_uint(lm));
}

__global__ void quant_out_kernel(const float* __restrict__ x,
                                 const unsigned* __restrict__ amax,
                                 float* __restrict__ y) {
  long i = (long)blockIdx.x * blockDim.x + threadIdx.x;
  if (i >= NELEM) return;
  float s = __uint_as_float(amax[0]) * (1.0f / 127.0f);
  y[i] = fminf(fmaxf(rintf(x[i] / s), -128.f), 127.f) * s;
}

// ---------------------------------------------------------------------------
extern "C" void kernel_launch(void* const* d_in, const int* in_sizes, int n_in,
                              void* d_out, int out_size, void* d_ws, size_t ws_size,
                              hipStream_t stream) {
  const float* hidden = (const float*)d_in[0];
  // d_in[1] = attention_mask (pure causal additive mask; computed in-kernel)
  const float* W_attn = (const float*)d_in[2];
  const float* b_attn = (const float*)d_in[3];
  const float* W_proj = (const float*)d_in[4];
  const float* b_proj = (const float*)d_in[5];
  float* out = (float*)d_out;
  float* present_k = out + NELEM;           // [B,H,S,hd]
  float* present_v = present_k + NELEM;

  char* ws = (char*)d_ws;                   // all regions fully rewritten per call
  float*    qkv  = (float*)(ws);                       // 4096x3072 f32 (48 MiB)
  float*    attn = (float*)(ws + 50331648);            // 4096x1024 f32 (16 MiB)
  float*    qout = (float*)(ws + 67108864);            // 4096x1024 f32 (16 MiB)
  __bf16*   kbf  = (__bf16*)(ws + 83886080);           // [B,H,S,hd] bf16 (8 MiB)
  __bf16*   vbf  = (__bf16*)(ws + 92274688);           // [B,H,S,hd] bf16 (8 MiB)
  unsigned* amax = (unsigned*)(ws + 100663296);        // {k, v, out}

  init_amax_kernel<<<1, 32, 0, stream>>>(amax);

  // qkv = hidden @ W_attn^T + b_attn
  gemm_bf16_wmma<<<dim3(E3 / 64, MROWS / 128), 128, 0, stream>>>(
      hidden, W_attn, b_attn, qkv, MROWS, E3, D);

  amax2d_kernel<<<1024, 256, 0, stream>>>(qkv + D,     E3, D, (int)NELEM, amax + 0);
  amax2d_kernel<<<1024, 256, 0, stream>>>(qkv + 2 * D, E3, D, (int)NELEM, amax + 1);

  quant_kv_kernel<<<(int)(NELEM / 256), 256, 0, stream>>>(
      qkv, amax, present_k, present_v, kbf, vbf);

  attn_kernel<<<dim3(S / 64, Bz * H), 128, 0, stream>>>(
      qkv, kbf, vbf, attn, amax + 2);

  quant_out_kernel<<<(int)(NELEM / 256), 256, 0, stream>>>(attn, amax + 2, qout);

  // out = qout @ W_proj^T + b_proj
  gemm_bf16_wmma<<<dim3(D / 64, MROWS / 128), 128, 0, stream>>>(
      qout, W_proj, b_proj, out, MROWS, D, D);
}
